// GPREDecoder_58213986730273
// MI455X (gfx1250) — matrix operand order repaired and checked by reading, not compiled, not run on hardware
//
#include <hip/hip_runtime.h>

// ---------------------------------------------------------------------------
// GPREDecoder loss for MI455X (gfx1250, wave32, WMMA bf16 16x16x32)
// B=8 S=1024 HID=1024 LAB=64 NAH=16 HD=68 P=64
// Maps: 0..15 ent (b=mi>>1,h=mi&1, RoPE+tril), 16..23 head, 24..31 tail
// Staging: TDM tensor_load_to_lds (TENSORcnt) for Q/K logit tiles,
//          async-to-LDS ops (ASYNCcnt) for the projection GEMM tiles.
// ---------------------------------------------------------------------------

#define INFC   1000000000000.0f
#define SCALEF 0.121267812518166f   // 1/sqrt(68)
#define SQ     1024
#define DIN    1088                 // HID+LAB
#define NJ     544                  // 272 + 136 + 136 stacked projection rows
#define HDP    96                   // HD=68 padded to 3*32 for bf16 WMMA K-chunks
#define NMAP   32

typedef float  v8f   __attribute__((ext_vector_type(8)));
typedef __bf16 bf8   __attribute__((ext_vector_type(8)));
typedef __bf16 v16bf __attribute__((ext_vector_type(16)));
typedef unsigned int u32x4 __attribute__((ext_vector_type(4)));
typedef int          i32x8 __attribute__((ext_vector_type(8)));
typedef int          i32x4 __attribute__((ext_vector_type(4)));

static __device__ __forceinline__ unsigned short f2bf(float f) {
  unsigned int u = __float_as_uint(f);
  unsigned int r = (u + 0x7fffu + ((u >> 16) & 1u)) >> 16;   // RNE
  return (unsigned short)r;
}
static __device__ __forceinline__ float bf2f(unsigned short h) {
  return __uint_as_float(((unsigned int)h) << 16);
}

static __device__ __forceinline__ v16bf frag_cat(bf8 lo, bf8 hi) {
  return __builtin_shufflevector(lo, hi, 0,1,2,3,4,5,6,7,8,9,10,11,12,13,14,15);
}

// --- gfx1250 async global->LDS copies (no VGPR return, tracked by ASYNCcnt) --
static __device__ __forceinline__ unsigned lds_off32(const void* p) {
  // generic LDS pointer: addr[31:0] is the LDS byte address (ISA 10.2)
  return (unsigned)(unsigned long long)p;
}
static __device__ __forceinline__ void async_ld_b128(unsigned lds, const void* g) {
  asm volatile("global_load_async_to_lds_b128 %0, %1, off"
               :: "v"(lds), "v"((unsigned long long)g) : "memory");
}
static __device__ __forceinline__ void async_ld_b64(unsigned lds, const void* g) {
  asm volatile("global_load_async_to_lds_b64 %0, %1, off"
               :: "v"(lds), "v"((unsigned long long)g) : "memory");
}
static __device__ __forceinline__ void wait_async0() {
#if __has_builtin(__builtin_amdgcn_s_wait_asynccnt)
  __builtin_amdgcn_s_wait_asynccnt(0);
#else
  asm volatile("s_wait_asynccnt 0x0" ::: "memory");
#endif
}

// --- gfx1250 Tensor Data Mover: 2D tile global->LDS (tracked by TENSORcnt) ---
#if __has_builtin(__builtin_amdgcn_tensor_load_to_lds)
#define HAVE_TDM 1
// D# layout per CDNA5 ISA 8.3/8.4. 2D tensor: groups 2/3 unused (zero).
// This toolchain's builtin takes 6 args: (v4u g0, v8i g1, v4i, v4i, v8i, i32 cpol)
static __device__ __forceinline__ void tdm_load_2d(
    unsigned lds_addr, const void* gaddr,
    unsigned tile_d0, unsigned tile_d1,          // elements, rows
    unsigned tensor_d0, unsigned tensor_d1,      // elements, rows
    unsigned stride0) {                          // elements between rows
  const unsigned long long ga = (unsigned long long)gaddr;
  u32x4 g0;
  g0.x = 1u;                                     // count=1 (valid descriptor)
  g0.y = lds_addr;                               // lds_addr[31:0]
  g0.z = (unsigned)ga;                           // global_addr[31:0]
  g0.w = (unsigned)(ga >> 32) | (2u << 30);      // global_addr[56:32] | type=2
  i32x8 g1;
  g1[0] = (int)(1u << 16);                       // data_size=1 (2 bytes)
  g1[1] = (int)((tensor_d0 & 0xffffu) << 16);    // tensor_dim0[15:0]
  g1[2] = (int)(((tensor_d0 >> 16) & 0xffffu) |  // tensor_dim0[31:16]
                ((tensor_d1 & 0xffffu) << 16));  // tensor_dim1[15:0]
  g1[3] = (int)(((tensor_d1 >> 16) & 0xffffu) |  // tensor_dim1[31:16]
                ((tile_d0 & 0xffffu) << 16));    // tile_dim0
  g1[4] = (int)(tile_d1 & 0xffffu);              // tile_dim1 (tile_dim2=0)
  g1[5] = (int)stride0;                          // tensor_dim0_stride[31:0]
  g1[6] = 0;                                     // stride0[47:32] | stride1[15:0]
  g1[7] = 0;                                     // stride1[47:16]
  const i32x4 gz4 = {0, 0, 0, 0};
  const i32x8 gz8 = {0, 0, 0, 0, 0, 0, 0, 0};
  __builtin_amdgcn_tensor_load_to_lds(g0, g1, gz4, gz4, gz8, 0);
}
static __device__ __forceinline__ void wait_tensor0() {
  __builtin_amdgcn_s_wait_tensorcnt((short)0);
}
#endif

// ---------------- kernel 0: zero scalars ----------------
__global__ void k_zero(float* negsum, float* out) {
  int t = threadIdx.x;
  if (t < NMAP) negsum[t] = 0.0f;
  if (t == 0)   out[0]    = 0.0f;
}

// ---------------- kernel 1: build X (bf16, 8192 x 1088) ----------------
__global__ void k_build_x(const float* __restrict__ hidden,
                          const int*   __restrict__ labels,
                          const float* __restrict__ ent_emb,
                          unsigned short* __restrict__ X) {
  const int m   = blockIdx.x;             // 0..8191  (b*1024+s)
  const int lab = labels[m];
  for (int d = threadIdx.x; d < DIN; d += blockDim.x) {
    float v = (d < 1024) ? hidden[(size_t)m * 1024 + d]
                         : ent_emb[lab * 64 + (d - 1024)];
    X[(size_t)m * DIN + d] = f2bf(v);
  }
}

// ---------------- kernel 2: build stacked W (bf16, 544 x 1088) ----------------
__global__ void k_build_w(const float* __restrict__ We,
                          const float* __restrict__ Wh,
                          const float* __restrict__ Wt,
                          unsigned short* __restrict__ W) {
  const int j = blockIdx.x;               // 0..543
  const float* src = (j < 272) ? (We + (size_t)j * DIN)
                   : (j < 408) ? (Wh + (size_t)(j - 272) * DIN)
                               : (Wt + (size_t)(j - 408) * DIN);
  for (int d = threadIdx.x; d < DIN; d += blockDim.x)
    W[(size_t)j * DIN + d] = f2bf(src[d]);
}

// ---------------- kernel 3: projection GEMM (8192x544 = X @ Wcat^T) -----------
// block = 256 thr (8 waves); tile 128(M) x 32(N); K loop 34 chunks of 32.
__global__ void k_proj_gemm(const unsigned short* __restrict__ X,
                            const unsigned short* __restrict__ W,
                            float* __restrict__ P) {
  const int n0  = blockIdx.x * 32;
  const int m0  = blockIdx.y * 128;
  const int tid  = threadIdx.x;
  const int wv   = tid >> 5;
  const int lane = tid & 31;
  const int lh   = lane & 15;
  const bool lo16 = lane < 16;

  __shared__ __align__(16) unsigned short sX[128 * 32];
  __shared__ __align__(16) unsigned short sW[32 * 32];

  // per-thread staging addresses (advance k0 by pointer bump)
  const int xrow  = tid >> 1;
  const int xpart = (tid & 1) * 16;
  const char* xsrc = (const char*)(X + (size_t)(m0 + xrow) * DIN + xpart);
  const unsigned xdst = lds_off32(sX + xrow * 32 + xpart);
  const int wrow  = tid >> 3;
  const int wpart = (tid & 7) * 4;
  const char* wsrc = (const char*)(W + (size_t)(n0 + wrow) * DIN + wpart);
  const unsigned wdst = lds_off32(sW + wrow * 32 + wpart);

  v8f acc0 = (v8f)0.0f, acc1 = (v8f)0.0f;

  for (int kc = 0; kc < 34; ++kc) {
    __syncthreads();                       // WAR: previous iteration's DS reads
    async_ld_b128(xdst,      xsrc + (size_t)kc * 64);
    async_ld_b128(xdst + 16, xsrc + (size_t)kc * 64 + 16);
    async_ld_b64 (wdst,      wsrc + (size_t)kc * 64);
    wait_async0();
    __syncthreads();

    // A fragment: rows wv*16 .. +15
    const unsigned short* ap = sX + (wv * 16 + lh) * 32 + (lo16 ? 0 : 16);
    v16bf a = frag_cat(*(const bf8*)ap, *(const bf8*)(ap + 8));

    // B fragments: two 16-col subtiles
    {
      const unsigned short* bp = sW + lh * 32 + (lo16 ? 0 : 16);
      v16bf b = frag_cat(*(const bf8*)bp, *(const bf8*)(bp + 8));
      acc0 = __builtin_amdgcn_wmma_f32_16x16x32_bf16(false, a, false, b,
                                                     (short)0, acc0, false, false);
    }
    {
      const unsigned short* bp = sW + (16 + lh) * 32 + (lo16 ? 0 : 16);
      v16bf b = frag_cat(*(const bf8*)bp, *(const bf8*)(bp + 8));
      acc1 = __builtin_amdgcn_wmma_f32_16x16x32_bf16(false, a, false, b,
                                                     (short)0, acc1, false, false);
    }
  }

  const int mbase = m0 + wv * 16 + (lo16 ? 0 : 8);
  const int nidx0 = n0 + lh;
  const int nidx1 = n0 + 16 + lh;
#pragma unroll
  for (int r = 0; r < 8; ++r) {
    P[(size_t)(mbase + r) * NJ + nidx0] = acc0[r];
    P[(size_t)(mbase + r) * NJ + nidx1] = acc1[r];
  }
}

// ---------------- kernel 4: bias + RoPE + pad, write bf16 Q/K -----------------
// grid (1024 s, 32 mi), 192 threads: t<96 -> q col t, t>=96 -> k col t-96
__global__ void k_postproc(const float* __restrict__ P,
                           const float* __restrict__ be,
                           const float* __restrict__ bh,
                           const float* __restrict__ bt,
                           unsigned short* __restrict__ Qb,
                           unsigned short* __restrict__ Kb) {
  const int s  = blockIdx.x;
  const int mi = blockIdx.y;
  const int t  = threadIdx.x;
  const int side = (t >= 96) ? 1 : 0;
  const int cc   = side ? (t - 96) : t;

  int b, base_j;
  bool rope;
  if (mi < 16)      { b = mi >> 1; base_j = (mi & 1) * 136 + side * 68; rope = true;  }
  else if (mi < 24) { b = mi - 16; base_j = 272 + side * 68;            rope = false; }
  else              { b = mi - 24; base_j = 408 + side * 68;            rope = false; }

  const size_t m = (size_t)b * SQ + s;
  unsigned short outv = 0;
  if (cc < 68) {
    const int j  = base_j + cc;
    const float bias  = (j < 272) ? be[j] : (j < 408) ? bh[j - 272] : bt[j - 408];
    float v = P[m * NJ + j] + bias;
    if (rope) {
      const int jp = base_j + (cc ^ 1);
      const float biasp = (jp < 272) ? be[jp] : (jp < 408) ? bh[jp - 272] : bt[jp - 408];
      const float vp  = P[m * NJ + jp] + biasp;
      const int   i   = cc >> 1;
      const float inv = __powf(10000.0f, -2.0f * (float)i / 68.0f);
      const float ang = (float)s * inv;
      const float sn = __sinf(ang), cs = __cosf(ang);
      v = ((cc & 1) == 0) ? (v * cs - vp * sn) : (v * cs + vp * sn);
    }
    outv = f2bf(v);
  }
  unsigned short* dst = side ? Kb : Qb;
  dst[((size_t)mi * SQ + s) * HDP + cc] = outv;
}

// ---------------- kernel 5: fused logits + exp-sum (mcce negative region) -----
// grid 2048 = 32 maps * 64 tiles (8x8 of 128x128); 256 thr = 8 waves.
__global__ void k_logits_neg(const unsigned short* __restrict__ Qb,
                             const unsigned short* __restrict__ Kb,
                             const float* __restrict__ mask,
                             float* __restrict__ negsum) {
  const int blk  = blockIdx.x;
  const int mi   = blk >> 6;
  const int tile = blk & 63;
  const int tm   = tile >> 3, tn = tile & 7;
  const bool is_ent = (mi < 16);
  if (is_ent && tm > tn) return;            // strictly-lower tiles: exp == 0

  const int b = is_ent ? (mi >> 1) : (mi < 24 ? mi - 16 : mi - 24);
  const int tid  = threadIdx.x;
  const int wv   = tid >> 5;
  const int lane = tid & 31;
  const int lh   = lane & 15;
  const bool lo16 = lane < 16;

  __shared__ __align__(16) unsigned short sQ[128 * HDP];
  __shared__ __align__(16) unsigned short sK[128 * HDP];
  __shared__ float swav[8];

#ifdef HAVE_TDM
  // TDM: wave 0 DMAs the 128x96 Q tile, wave 1 the K tile (EXEC ignored,
  // so only the branched-in waves issue). One instruction per 24 KB tile.
  if (tid < 64) {
    const int side = wv;                    // 0 = Q, 1 = K
    const unsigned short* src = side
        ? (Kb + ((size_t)mi * SQ + tn * 128) * HDP)
        : (Qb + ((size_t)mi * SQ + tm * 128) * HDP);
    const unsigned dst = lds_off32(side ? sK : sQ);
    tdm_load_2d(dst, src, /*tile*/ HDP, 128, /*tensor*/ HDP, SQ, /*stride*/ HDP);
    wait_tensor0();
  }
#else
  { // async staging: 2 threads/row, 48 halfs (96B = 6 x b128) per thread/tile
    const int row  = tid >> 1;
    const int part = (tid & 1) * 48;
    const char* qs = (const char*)(Qb + ((size_t)mi * SQ + tm * 128 + row) * HDP + part);
    const char* ks = (const char*)(Kb + ((size_t)mi * SQ + tn * 128 + row) * HDP + part);
    const unsigned qd = lds_off32(sQ + row * HDP + part);
    const unsigned kd = lds_off32(sK + row * HDP + part);
#pragma unroll
    for (int i = 0; i < 6; ++i) {
      async_ld_b128(qd + i * 16, qs + i * 16);
      async_ld_b128(kd + i * 16, ks + i * 16);
    }
    wait_async0();
  }
#endif
  __syncthreads();

  v8f acc[8];
#pragma unroll
  for (int i = 0; i < 8; ++i) acc[i] = (v8f)0.0f;

#pragma unroll
  for (int kc = 0; kc < 3; ++kc) {
    const unsigned short* ap = sQ + (wv * 16 + lh) * HDP + kc * 32 + (lo16 ? 0 : 16);
    v16bf a = frag_cat(*(const bf8*)ap, *(const bf8*)(ap + 8));
#pragma unroll
    for (int sub = 0; sub < 8; ++sub) {
      const unsigned short* bp = sK + (sub * 16 + lh) * HDP + kc * 32 + (lo16 ? 0 : 16);
      v16bf bb = frag_cat(*(const bf8*)bp, *(const bf8*)(bp + 8));
      acc[sub] = __builtin_amdgcn_wmma_f32_16x16x32_bf16(false, a, false, bb,
                                                         (short)0, acc[sub], false, false);
    }
  }

  // exp-sum epilogue (C layout: vgpr r -> M = r + (lane>=16)*8, N = lane&15)
  float ssum = 0.0f;
  const int mbase = tm * 128 + wv * 16 + (lo16 ? 0 : 8);
#pragma unroll
  for (int sub = 0; sub < 8; ++sub) {
    const int   n   = tn * 128 + sub * 16 + lh;
    const float pad = mask[(size_t)b * SQ + n];
#pragma unroll
    for (int r = 0; r < 8; ++r) {
      const int m = mbase + r;
      float v = acc[sub][r] * SCALEF;
      v = v * pad - (1.0f - pad) * INFC;
      if (is_ent && m > n) v -= INFC;       // tril(-1) mask
      ssum += __expf(v);
    }
  }
#pragma unroll
  for (int off = 16; off > 0; off >>= 1) ssum += __shfl_xor(ssum, off, 32);
  if (lane == 0) swav[wv] = ssum;
  __syncthreads();
  if (tid == 0) {
    float tot = 0.0f;
#pragma unroll
    for (int i = 0; i < 8; ++i) tot += swav[i];
    atomicAdd(&negsum[mi], tot);
  }
}

// ---------------- kernel 6: label corrections + final loss --------------------
// 32 blocks (one per map), 64 threads (one per label p)
__global__ void k_finish(const unsigned short* __restrict__ Qb,
                         const unsigned short* __restrict__ Kb,
                         const float* __restrict__ mask,
                         const int* __restrict__ gte,
                         const int* __restrict__ gth,
                         const int* __restrict__ gtt,
                         const float* __restrict__ negsum,
                         float* __restrict__ out) {
  const int mi = blockIdx.x;
  const int p  = threadIdx.x;
  const bool is_ent = (mi < 16);
  const int b = is_ent ? (mi >> 1) : (mi < 24 ? mi - 16 : mi - 24);

  const int* g = is_ent ? (gte + ((size_t)mi * 64 + p) * 2)
               : (mi < 24) ? (gth + ((size_t)(mi - 16) * 64 + p) * 2)
                           : (gtt + ((size_t)(mi - 24) * 64 + p) * 2);
  const int r = g[0], c = g[1];
  const int flat = r * SQ + c;

  __shared__ int   sflat[64];
  __shared__ float spos[64], sneg[64];
  sflat[p] = flat;
  __syncthreads();

  const unsigned short* qr  = Qb + ((size_t)mi * SQ + r) * HDP;
  const unsigned short* kc_ = Kb + ((size_t)mi * SQ + c) * HDP;
  float dot = 0.0f;
#pragma unroll 4
  for (int d = 0; d < 68; ++d) dot += bf2f(qr[d]) * bf2f(kc_[d]);
  const float pad = mask[(size_t)b * SQ + c];
  float v = dot * SCALEF * pad - (1.0f - pad) * INFC;
  if (is_ent && r > c) v -= INFC;

  float pos = (flat != 0) ? __expf(-v) : 0.0f;

  bool first = true;
  for (int q = 0; q < p; ++q) if (sflat[q] == flat) first = false;
  float ncorr = (flat != 0 && first) ? __expf(v) : 0.0f;

  if (p == 0) { // exclude flat index 0 (logit at m=0,n=0) from the region sum
    const unsigned short* q0 = Qb + (size_t)mi * SQ * HDP;
    const unsigned short* k0 = Kb + (size_t)mi * SQ * HDP;
    float d0 = 0.0f;
    for (int d = 0; d < 68; ++d) d0 += bf2f(q0[d]) * bf2f(k0[d]);
    const float pad0 = mask[(size_t)b * SQ];
    ncorr += __expf(d0 * SCALEF * pad0 - (1.0f - pad0) * INFC);
  }

  spos[p] = pos;
  sneg[p] = ncorr;
  __syncthreads();
  if (p == 0) {
    float pt = 0.0f, nt = 0.0f;
    for (int i = 0; i < 64; ++i) { pt += spos[i]; nt += sneg[i]; }
    const float neg_total = negsum[mi] - nt + 2.0f;  // two exp(0) terms
    const float loss = logf(1.0f + pt) + logf(neg_total);
    atomicAdd(out, loss);
  }
}

// ---------------------------------------------------------------------------
extern "C" void kernel_launch(void* const* d_in, const int* in_sizes, int n_in,
                              void* d_out, int out_size, void* d_ws, size_t ws_size,
                              hipStream_t stream) {
  const float* hidden  = (const float*)d_in[0];
  const int*   labels  = (const int*)  d_in[1];
  const float* mask    = (const float*)d_in[2];
  const int*   gte     = (const int*)  d_in[3];
  const int*   gth     = (const int*)  d_in[4];
  const int*   gtt     = (const int*)  d_in[5];
  const float* ent_emb = (const float*)d_in[6];
  const float* W_ent   = (const float*)d_in[7];
  const float* b_ent   = (const float*)d_in[8];
  const float* W_head  = (const float*)d_in[9];
  const float* b_head  = (const float*)d_in[10];
  const float* W_tail  = (const float*)d_in[11];
  const float* b_tail  = (const float*)d_in[12];
  float* out = (float*)d_out;

  // workspace carving (256B aligned)
  char* w = (char*)d_ws;
  size_t off = 0;
  auto take = [&](size_t bytes) { char* p = w + off; off += (bytes + 255) & ~(size_t)255; return p; };
  unsigned short* X   = (unsigned short*)take((size_t)8192 * DIN * 2);
  unsigned short* Wc  = (unsigned short*)take((size_t)NJ * DIN * 2);
  float*          P   = (float*)         take((size_t)8192 * NJ * 4);
  unsigned short* Qb  = (unsigned short*)take((size_t)NMAP * SQ * HDP * 2);
  unsigned short* Kb  = (unsigned short*)take((size_t)NMAP * SQ * HDP * 2);
  float*          neg = (float*)         take(NMAP * 4);

  k_zero<<<1, 64, 0, stream>>>(neg, out);
  k_build_x<<<8192, 256, 0, stream>>>(hidden, labels, ent_emb, X);
  k_build_w<<<NJ, 256, 0, stream>>>(W_ent, W_head, W_tail, Wc);
  k_proj_gemm<<<dim3(NJ / 32, 8192 / 128), 256, 0, stream>>>(X, Wc, P);
  k_postproc<<<dim3(SQ, NMAP), 192, 0, stream>>>(P, b_ent, b_head, b_tail, Qb, Kb);
  k_logits_neg<<<NMAP * 64, 256, 0, stream>>>(Qb, Kb, mask, neg);
  k_finish<<<NMAP, 64, 0, stream>>>(Qb, Kb, mask, gte, gth, gtt, neg, out);
}
